// CopyGenerator_90915867721982
// MI455X (gfx1250) — matrix operand order repaired and checked by reading, not compiled, not run on hardware
//
#include <hip/hip_runtime.h>

// Problem constants (fixed by the reference harness)
constexpr int Vv = 50257;   // vocab
constexpr int Hh = 512;     // d_model
constexpr int Ss = 512;     // source len
constexpr int Bb = 8;       // batch
constexpr int Tt = 64;      // decode steps
constexpr int Mm = Tt * Bb; // 512 GEMM rows

// GEMM tiling: block = 64(M) x 128(N), K-chunk 32 (one bf16 WMMA K-step)
constexpr int BM = 64, BN = 128, KC = 32;
constexpr int NCH = Hh / KC;          // 16 K-chunks
constexpr int ROWB = 80;              // LDS row stride: 64B data + 16B TDM pad

typedef __attribute__((ext_vector_type(16))) __bf16       v16bf;
typedef __attribute__((ext_vector_type(8)))  float        v8f;
typedef __attribute__((ext_vector_type(4)))  unsigned int v4u;
typedef __attribute__((ext_vector_type(8)))  int          v8i;
typedef __attribute__((ext_vector_type(4)))  int          v4i;

// ---------------------------------------------------------------------------
// Kernel 0: W_gen [H,V] f32  ->  Wt [V,H] bf16  (tiled LDS transpose, 51 MB,
// L2-resident afterwards). Transposed layout makes the TDM deposit B tiles
// as [N][K] in LDS so B fragments are two contiguous ds_load_b128 per lane.
// ---------------------------------------------------------------------------
__global__ void cvt_w_bf16_t(const float* __restrict__ w, __bf16* __restrict__ wt) {
  __shared__ float tile[32][33];
  const int tx = threadIdx.x & 31;
  const int ty = threadIdx.x >> 5;          // 0..7
  const int v0 = blockIdx.x * 32;
  const int h0 = blockIdx.y * 32;
#pragma unroll
  for (int k = 0; k < 4; ++k) {
    int h = h0 + ty + 8 * k;
    int v = v0 + tx;
    tile[ty + 8 * k][tx] = (v < Vv) ? w[(size_t)h * Vv + v] : 0.f;
  }
  __syncthreads();
#pragma unroll
  for (int k = 0; k < 4; ++k) {
    int vv = ty + 8 * k;
    if (v0 + vv < Vv)
      wt[(size_t)(v0 + vv) * Hh + h0 + tx] = (__bf16)tile[tx][vv];
  }
}

// ---------------------------------------------------------------------------
// Kernel 1: mw[s*B+b] = memory[s,b,:] . W_prob   (one wave32 per (s,b))
// ---------------------------------------------------------------------------
__global__ void mw_kernel(const float* __restrict__ mem, const float* __restrict__ wp,
                          float* __restrict__ mw) {
  int gw   = (int)((blockIdx.x * blockDim.x + threadIdx.x) >> 5);
  int lane = threadIdx.x & 31;
  if (gw >= Ss * Bb) return;
  const float* row = mem + (size_t)gw * Hh;
  float sum = 0.f;
  for (int h = lane; h < Hh; h += 32) sum += row[h] * wp[h];
  for (int off = 16; off; off >>= 1) sum += __shfl_down(sum, off, 32);
  if (lane == 0) mw[gw] = sum;
}

// ---------------------------------------------------------------------------
// Kernel 2: prob[t*B+b] = sigmoid(attn[b,t,:] . mw[:,b] + b_prob)
// ---------------------------------------------------------------------------
__global__ void prob_kernel(const float* __restrict__ attn, const float* __restrict__ mw,
                            const float* __restrict__ bprob, float* __restrict__ prob) {
  int gw   = (int)((blockIdx.x * blockDim.x + threadIdx.x) >> 5);
  int lane = threadIdx.x & 31;
  if (gw >= Tt * Bb) return;
  int t = gw / Bb, b = gw % Bb;
  const float* arow = attn + ((size_t)b * Tt + t) * Ss;
  float sum = 0.f;
  for (int s = lane; s < Ss; s += 32) sum += arow[s] * mw[s * Bb + b];
  for (int off = 16; off; off >>= 1) sum += __shfl_down(sum, off, 32);
  if (lane == 0) prob[gw] = 1.f / (1.f + __expf(-(sum + bprob[0])));
}

// ---------------------------------------------------------------------------
// Kernel 3: C[M,V] = A[M,H] @ W[H,V] + b_gen, W given transposed bf16 Wt[V,H].
//   bf16 WMMA; B tiles via TDM (double-buffered, padded LDS rows);
//   block tile 64x128, 8 waves: 4(M) x 2(N), each wave 16x64 (4 accum tiles)
// ---------------------------------------------------------------------------
__global__ void gen_gemm(const float* __restrict__ A, const __bf16* __restrict__ Wt,
                         const float* __restrict__ bgen, float* __restrict__ C) {
  __shared__ alignas(16) char As[2][BM * ROWB];   // 2 x 5 KB, stride-80 rows
  __shared__ alignas(16) char Bs[2][BN * ROWB];   // 2 x 10 KB, TDM-filled

  const int tid  = threadIdx.x;
  const int lane = tid & 31;
  const int w    = tid >> 5;
  const int wm   = (w & 3) * 16;  // wave M offset inside block tile
  const int wn   = (w >> 2) * 64; // wave N offset inside block tile
  const int m0   = blockIdx.y * BM;
  const int n0   = blockIdx.x * BN;
  const int r    = lane & 15;
  const int half = lane >> 4;

  union Frag { uint4 q[2]; v16bf v; };

  // ---- TDM issue: tile [BN rows x KC cols] of Wt (row stride H), LDS-padded
  auto tdm_issue = [&](int k0, int buf) {
    unsigned long long ga =
        (unsigned long long)(const void*)(Wt + (size_t)n0 * Hh + k0);
    unsigned ldsOff = (unsigned)(unsigned long long)(const void*)&Bs[buf][0];
    // group0: count=1 | lds_addr | global_addr[56:0] | type=2
    v4u g0 = { 1u, ldsOff, (unsigned)ga,
               ((unsigned)((ga >> 32) & 0x01FFFFFFull)) | (2u << 30) };
    // group1: data_size=2B; pad_enable, pad every 16 dwords (64B) by 4 dwords
    //         tensor_dim0=H, tensor_dim1=V, tile_dim0=KC, tile_dim1=BN,
    //         tensor_dim0_stride=H   (rows beyond V read as zero)
    v8i g1 = { (int)((1u << 16) | (1u << 20) | (3u << 22) | (3u << 25)),
               (int)(((unsigned)Hh & 0xFFFFu) << 16),               // dim0 lo16
               (int)((((unsigned)Hh >> 16) & 0xFFFFu) |
                     (((unsigned)Vv & 0xFFFFu) << 16)),             // dim0 hi/dim1 lo
               (int)((((unsigned)Vv >> 16) & 0xFFFFu) |
                     (((unsigned)KC & 0xFFFFu) << 16)),             // dim1 hi/tile0
               (int)(unsigned)BN,                                   // tile1 (tile2=0)
               (int)(unsigned)Hh,                                   // stride0 lo32
               0, 0 };
    v4i g2 = { 0, 0, 0, 0 };
    v4i g3 = { 0, 0, 0, 0 };
    v8i g4 = { 0, 0, 0, 0, 0, 0, 0, 0 };
    __builtin_amdgcn_tensor_load_to_lds(g0, g1, g2, g3, g4, 0);
  };

  // ---- cooperative A chunk load + f32->bf16 (64 rows x 32 cols)
  auto a_stage = [&](int k0, int buf) {
    int row = tid >> 2;          // 0..63
    int cs  = (tid & 3) * 8;     // 8-element column segment
    const float* src = A + (size_t)(m0 + row) * Hh + k0 + cs;
    float4 f0 = *(const float4*)(src);
    float4 f1 = *(const float4*)(src + 4);
    union { __bf16 h[8]; uint4 q; } u;
    u.h[0] = (__bf16)f0.x; u.h[1] = (__bf16)f0.y;
    u.h[2] = (__bf16)f0.z; u.h[3] = (__bf16)f0.w;
    u.h[4] = (__bf16)f1.x; u.h[5] = (__bf16)f1.y;
    u.h[6] = (__bf16)f1.z; u.h[7] = (__bf16)f1.w;
    *(uint4*)(&As[buf][row * ROWB + cs * 2]) = u.q;
  };

  v8f acc[4];
#pragma unroll
  for (int i = 0; i < 4; ++i) acc[i] = (v8f)0.0f;

  // pipeline prologue: chunk 0 into buffer 0
  if (w == 0) tdm_issue(0, 0);
  a_stage(0, 0);

  for (int i = 0; i < NCH; ++i) {
    const int cur = i & 1, nxt = cur ^ 1;
    __syncthreads();                         // everyone done reading buf[nxt]
    if (i + 1 < NCH) {                       // prefetch chunk i+1
      if (w == 0) tdm_issue((i + 1) * KC, nxt);
      a_stage((i + 1) * KC, nxt);
    }
    if (w == 0) {                            // chunk i complete (i+1 may fly)
      if (i + 1 < NCH) __builtin_amdgcn_s_wait_tensorcnt(1);
      else             __builtin_amdgcn_s_wait_tensorcnt(0);
    }
    __syncthreads();                         // buf[cur] ready for all waves

    // A fragment 16x32 bf16: two b128 from padded row (ISA 7.12.2 layout)
    Frag af;
    {
      const char* ab = &As[cur][(wm + r) * ROWB + half * 16];
      af.q[0] = *(const uint4*)(ab);
      af.q[1] = *(const uint4*)(ab + 32);
    }
    // 4 WMMAs: B fragment = two 16B pieces of column ncol (k 8h..8h+7, 16+8h..)
#pragma unroll
    for (int nt = 0; nt < 4; ++nt) {
      Frag bf;
      const char* bb = &Bs[cur][(wn + nt * 16 + r) * ROWB + half * 16];
      bf.q[0] = *(const uint4*)(bb);
      bf.q[1] = *(const uint4*)(bb + 32);
      acc[nt] = __builtin_amdgcn_wmma_f32_16x16x32_bf16(
          false, af.v, false, bf.v, (short)0, acc[nt], false, false);
    }
  }

  // epilogue: C = acc + b_gen  (C f32 16x16 layout: VGPR j -> M = 8*half + j)
#pragma unroll
  for (int nt = 0; nt < 4; ++nt) {
#pragma unroll
    for (int j = 0; j < 8; ++j) {
      int row = m0 + wm + half * 8 + j;
      int col = n0 + wn + nt * 16 + r;
      if (col < Vv) C[(size_t)row * Vv + col] = acc[nt][j] + bgen[col];
    }
  }
}

// ---------------------------------------------------------------------------
// Kernel 4: per (t,b) row — LDS hash scatter of copy mass, Zc,
//           online softmax stats over gen logits (in d_out), final log-mix.
// ---------------------------------------------------------------------------
__global__ void final_rows(const int* __restrict__ src, const float* __restrict__ attn,
                           const float* __restrict__ probv, float* __restrict__ out) {
  constexpr int HSZ = 1024;
  __shared__ int   hkey[HSZ];
  __shared__ float hval[HSZ];
  __shared__ float rm[256], rs[256];
  __shared__ int   dcount;
  __shared__ float sZc, sMg, sZg;

  const int row = blockIdx.x;           // row = t*B + b
  const int t = row / Bb, b = row % Bb;
  const int tid = threadIdx.x;

  for (int i = tid; i < HSZ; i += 256) { hkey[i] = -1; hval[i] = 0.f; }
  if (tid == 0) dcount = 0;
  __syncthreads();

  // scatter-add attention mass, merging duplicate vocab ids in LDS
  for (int s = tid; s < Ss; s += 256) {
    int   v = src[s * Bb + b];
    float a = attn[((size_t)b * Tt + t) * Ss + s];
    unsigned h = ((unsigned)v * 2654435761u) & (HSZ - 1);
    for (;;) {
      int prev = atomicCAS(&hkey[h], -1, v);
      if (prev == -1 || prev == v) {
        if (prev == -1) atomicAdd(&dcount, 1);
        atomicAdd(&hval[h], a);
        break;
      }
      h = (h + 1) & (HSZ - 1);
    }
  }
  __syncthreads();

  // Zc = (V - distinct)*exp(0) + sum_hit exp(c_v)
  float zc = 0.f;
  for (int i = tid; i < HSZ; i += 256)
    if (hkey[i] != -1) zc += __expf(hval[i]);
  rs[tid] = zc; __syncthreads();
  for (int off = 128; off; off >>= 1) {
    if (tid < off) rs[tid] += rs[tid + off];
    __syncthreads();
  }
  if (tid == 0) sZc = rs[0] + (float)(Vv - dcount);

  // online softmax (max, sum-exp) over gen logits stored in out[]
  float m = -3.4e38f, ssum = 0.f;
  const float* g = out + (size_t)row * Vv;
  for (int v = tid; v < Vv; v += 256) {
    float x = g[v];
    if (x > m) { ssum = ssum * __expf(m - x) + 1.f; m = x; }
    else       { ssum += __expf(x - m); }
  }
  rm[tid] = m; rs[tid] = ssum; __syncthreads();
  for (int off = 128; off; off >>= 1) {
    if (tid < off) {
      float m2 = rm[tid + off], s2 = rs[tid + off];
      if (m2 > rm[tid]) { rs[tid] = rs[tid] * __expf(rm[tid] - m2) + s2; rm[tid] = m2; }
      else              { rs[tid] += s2 * __expf(m2 - rm[tid]); }
    }
    __syncthreads();
  }
  if (tid == 0) { sMg = rm[0]; sZg = rs[0]; }
  __syncthreads();

  const float p  = probv[row];
  const float ca = p / sZg;
  const float cb = (1.f - p) / sZc;
  const float mg = sMg;
  float* o = out + (size_t)row * Vv;
  for (int v = tid; v < Vv; v += 256) {
    unsigned h = ((unsigned)v * 2654435761u) & (HSZ - 1);
    float ec = 1.f;                      // unhit slot: exp(0)
    for (;;) {
      int k = hkey[h];
      if (k == v)  { ec = __expf(hval[h]); break; }
      if (k == -1) break;
      h = (h + 1) & (HSZ - 1);
    }
    float gv = o[v];
    o[v] = __logf(ca * __expf(gv - mg) + cb * ec);
  }
}

// ---------------------------------------------------------------------------
extern "C" void kernel_launch(void* const* d_in, const int* in_sizes, int n_in,
                              void* d_out, int out_size, void* d_ws, size_t ws_size,
                              hipStream_t stream) {
  const int*   src   = (const int*)d_in[0];
  const float* dec   = (const float*)d_in[1];   // [T,B,H]
  const float* attn  = (const float*)d_in[2];   // [B,T,S]
  const float* mem   = (const float*)d_in[3];   // [S,B,H]
  const float* wgen  = (const float*)d_in[4];   // [H,V]
  const float* bgen  = (const float*)d_in[5];   // [V]
  const float* wprob = (const float*)d_in[6];   // [H]
  const float* bprob = (const float*)d_in[7];   // [1]
  float* out = (float*)d_out;

  // workspace: Wt (V*H bf16 = 51.5 MB) | mw (S*B f32) | prob (T*B f32)
  char*   ws   = (char*)d_ws;
  __bf16* wt   = (__bf16*)ws;
  float*  mw   = (float*)(ws + (size_t)Vv * Hh * 2);
  float*  prob = (float*)(ws + (size_t)Vv * Hh * 2 + (size_t)Ss * Bb * 4);

  dim3 gt((Vv + 31) / 32, Hh / 32);             // 1571 x 16
  cvt_w_bf16_t<<<gt, 256, 0, stream>>>(wgen, wt);
  mw_kernel  <<<(Ss * Bb) / 8, 256, 0, stream>>>(mem, wprob, mw);
  prob_kernel<<<(Tt * Bb) / 8, 256, 0, stream>>>(attn, mw, bprob, prob);

  dim3 g((Vv + BN - 1) / BN, Mm / BM);          // 393 x 8 blocks
  gen_gemm<<<g, 256, 0, stream>>>(dec, wt, bgen, out);

  final_rows<<<Mm, 256, 0, stream>>>(src, attn, prob, out);
}
